// SequenceAxialAttention_37383395345103
// MI455X (gfx1250) — matrix operand order repaired and checked by reading, not compiled, not run on hardware
//
#include <hip/hip_runtime.h>
#include <hip/hip_bf16.h>
#include <math.h>

// ---------------------------------------------------------------------------
// Types
// ---------------------------------------------------------------------------
typedef _Float16 half_t;
typedef __attribute__((ext_vector_type(16))) _Float16 v16h;
typedef __attribute__((ext_vector_type(8)))  _Float16 v8h;
typedef __attribute__((ext_vector_type(8)))  float    v8f;

#define WMMA_F16(a, b, c) \
    __builtin_amdgcn_wmma_f32_16x16x32_f16(false, (a), false, (b), (short)0, (c), false, false)

// Assemble a v16h A/B fragment from two 16-byte LDS chunks (2x ds_read_b128).
static __device__ __forceinline__ v16h ld2(const half_t* p0, const half_t* p1) {
    v8h a = *(const v8h*)p0;
    v8h b = *(const v8h*)p1;
    v16h r;
#pragma unroll
    for (int i = 0; i < 8; ++i) { r[i] = a[i]; r[i + 8] = b[i]; }
    return r;
}

static __device__ __forceinline__ v8f vzero8() {
    v8f z = {0.f, 0.f, 0.f, 0.f, 0.f, 0.f, 0.f, 0.f};
    return z;
}

// CDNA5 async global->LDS copy (ASYNCcnt-tracked, no VGPR data round-trip).
// GVS addressing: mem = saddr(64) + vaddr(32, bytes); vdst VGPR holds LDS byte
// address. Tracked with ASYNCcnt; waited with s_wait_asynccnt.
static __device__ __forceinline__ void async_b128(unsigned lds_addr,
                                                  unsigned voff,
                                                  unsigned long long sbase) {
    asm volatile("global_load_async_to_lds_b128 %0, %1, %2"
                 :: "v"(lds_addr), "v"(voff), "s"(sbase)
                 : "memory");
}
static __device__ __forceinline__ void wait_async0() {
    asm volatile("s_wait_asynccnt 0x0" ::: "memory");
}

// Problem constants
#define CDIM   512
#define NHEAD  8
#define HD     64
#define TDIM   64
#define NTOK   32768       // B * H * W = 8 * 64 * 64
#define FFDIM  2048

// ---------------------------------------------------------------------------
// Weight convert+transpose: W[k][n] f32 -> Wt[n][k] f16  (tiled via LDS)
// grid = (K/32, N/32), 256 threads (32x8)
// ---------------------------------------------------------------------------
__global__ void wcvt_kernel(const float* __restrict__ W, half_t* __restrict__ Wt,
                            int K, int N) {
    __shared__ float t[32][33];
    const int k0 = blockIdx.x * 32, n0 = blockIdx.y * 32;
    const int tx = threadIdx.x & 31, ty = threadIdx.x >> 5;
#pragma unroll
    for (int i = 0; i < 4; ++i)
        t[ty + i * 8][tx] = W[(size_t)(k0 + ty + i * 8) * N + n0 + tx];
    __syncthreads();
#pragma unroll
    for (int i = 0; i < 4; ++i)
        Wt[(size_t)(n0 + ty + i * 8) * K + k0 + tx] = (half_t)t[tx][ty + i * 8];
}

// ---------------------------------------------------------------------------
// LayerNorm: f32 tokens -> f16, optional (h,w) swap of the output row index
// grid = NTOK blocks, 128 threads
// ---------------------------------------------------------------------------
__global__ void ln_kernel(const float* __restrict__ X,
                          const float* __restrict__ gam,
                          const float* __restrict__ bet,
                          half_t* __restrict__ out, int swap) {
    const int t   = blockIdx.x;
    const int tid = threadIdx.x;
    __shared__ float red[128];
    __shared__ float red2[128];

    const float* xp = X + (size_t)t * CDIM;
    float loc[4];
    float s = 0.f, s2 = 0.f;
#pragma unroll
    for (int i = 0; i < 4; ++i) {
        float v = xp[tid + i * 128];
        loc[i] = v; s += v; s2 += v * v;
    }
    red[tid] = s; red2[tid] = s2;
    __syncthreads();
    for (int st = 64; st > 0; st >>= 1) {
        if (tid < st) { red[tid] += red[tid + st]; red2[tid] += red2[tid + st]; }
        __syncthreads();
    }
    const float mean = red[0] * (1.0f / CDIM);
    const float var  = red2[0] * (1.0f / CDIM) - mean * mean;
    const float rs   = rsqrtf(var + 1e-4f);

    int r = t;
    if (swap) {  // token (b, i1, i2) -> row (b, i2, i1)
        int b = t >> 12, i1 = (t >> 6) & 63, i2 = t & 63;
        r = (b << 12) + (i2 << 6) + i1;
    }
    half_t* op = out + (size_t)r * CDIM;
#pragma unroll
    for (int i = 0; i < 4; ++i) {
        int c = tid + i * 128;
        op[c] = (half_t)((loc[i] - mean) * rs * gam[c] + bet[c]);
    }
}

// ---------------------------------------------------------------------------
// WMMA GEMM:  Out = epilogue(A[f16, MxK] * Bt[f16, NxK]^T + bias)
// Block: 256 threads = 8 waves (4 in M x 2 in N). Tile 128x128x32.
// Double-buffered LDS, filled with GLOBAL_LOAD_ASYNC_TO_LDS_B128.
// MODE 0: store f16 (LDS-staged coalesced b128 stores)
// MODE 1: store f16, exact GELU (LDS-staged)
// MODE 2: store f32, scatter with (h,w) swap (row-attn proj), '='
// MODE 3: store f32, identity rows, '+='  (col-attn proj / ff2 residual)
// ---------------------------------------------------------------------------
template <int MODE>
__global__ __launch_bounds__(256) void gemm_kernel(
        const half_t* __restrict__ A,
        const half_t* __restrict__ Bt,
        const float* __restrict__ bias,
        void* __restrict__ Out,
        int M, int N, int K) {
    constexpr int BM = 128, BN = 128, BK = 32;
    // [0,8K)=As0 [8K,16K)=As1 [16K,24K)=Bs0 [24K,32K)=Bs1 ; epilogue C-tile 34K
    __shared__ __align__(16) char smem[34816];

    const int tid  = threadIdx.x;
    const int lane = tid & 31;
    const int wid  = tid >> 5;
    const int wm   = wid & 3;
    const int wn   = wid >> 2;
    const int l15  = lane & 15;
    const int lhi  = lane >> 4;
    const int block_m = blockIdx.y * BM;
    const int block_n = blockIdx.x * BN;

    const int row = tid >> 1;           // 0..127 (staging row)
    const int off = (tid & 1) * 16;     // halves

    const unsigned smem_base = (unsigned)(size_t)(&smem[0]);
    const unsigned long long abase = (unsigned long long)(size_t)A;
    const unsigned long long bbase = (unsigned long long)(size_t)Bt;
    const unsigned stg = (unsigned)(row * (BK * 2) + off * 2);  // byte off in tile

    v8f acc[2][4];
#pragma unroll
    for (int mt = 0; mt < 2; ++mt)
#pragma unroll
        for (int nt = 0; nt < 4; ++nt) acc[mt][nt] = vzero8();

    // ---- async fill of one double-buffer slot ----
    auto fill = [&](int buf, int k0) {
        const unsigned la = smem_base + (unsigned)(buf * 8192) + stg;
        const unsigned va = (unsigned)(((unsigned)(block_m + row) * (unsigned)K + k0 + off) * 2u);
        async_b128(la,      va,      abase);
        async_b128(la + 16, va + 16, abase);
        const unsigned lb = smem_base + 16384u + (unsigned)(buf * 8192) + stg;
        const unsigned vb = (unsigned)(((unsigned)(block_n + row) * (unsigned)K + k0 + off) * 2u);
        async_b128(lb,      vb,      bbase);
        async_b128(lb + 16, vb + 16, bbase);
    };

    fill(0, 0);
    int buf = 0;
    for (int k0 = 0; k0 < K; k0 += BK, buf ^= 1) {
        wait_async0();          // our fills of `buf` have landed in LDS
        __syncthreads();        // everyone's fills landed; prior reads done
        if (k0 + BK < K) fill(buf ^ 1, k0 + BK);

        const half_t* Asb = (const half_t*)(smem + buf * 8192);
        const half_t* Bsb = (const half_t*)(smem + 16384 + buf * 8192);

        v16h afrag[2], bfrag[4];
#pragma unroll
        for (int mt = 0; mt < 2; ++mt) {
            const half_t* p = Asb + (wm * 32 + mt * 16 + l15) * BK + lhi * 8;
            afrag[mt] = ld2(p, p + 16);     // K: {c0..c0+7, c0+16..c0+23}
        }
#pragma unroll
        for (int nt = 0; nt < 4; ++nt) {
            const half_t* p = Bsb + (wn * 64 + nt * 16 + l15) * BK + lhi * 8;
            bfrag[nt] = ld2(p, p + 16);     // K: {c0..c0+7, c0+16..c0+23}
        }
#pragma unroll
        for (int mt = 0; mt < 2; ++mt)
#pragma unroll
            for (int nt = 0; nt < 4; ++nt)
                acc[mt][nt] = WMMA_F16(afrag[mt], bfrag[nt], acc[mt][nt]);
    }

    // ---------------- epilogue ----------------
    if (MODE == 0 || MODE == 1) {
        __syncthreads();                       // done with A/B tiles; reuse smem
        half_t* Ct = (half_t*)smem;            // 128 x 128, stride 136 halves
#pragma unroll
        for (int mt = 0; mt < 2; ++mt) {
#pragma unroll
            for (int nt = 0; nt < 4; ++nt) {
                const int cc = wn * 64 + nt * 16 + l15;
                const float bv = bias[block_n + cc];
                const int rb = wm * 32 + mt * 16 + lhi * 8;
#pragma unroll
                for (int e = 0; e < 8; ++e) {
                    float val = acc[mt][nt][e] + bv;
                    if (MODE == 1)
                        val = 0.5f * val * (1.0f + erff(val * 0.70710678118654752f));
                    Ct[(rb + e) * 136 + cc] = (half_t)val;
                }
            }
        }
        __syncthreads();
        // coalesced write-out: 128 rows x 128 halves, 16B chunks
#pragma unroll
        for (int i = 0; i < 8; ++i) {
            int idx = tid + i * 256;
            int r = idx >> 4, c = (idx & 15) * 8;
            *(float4*)((half_t*)Out + (size_t)(block_m + r) * N + block_n + c) =
                *(const float4*)&Ct[r * 136 + c];
        }
    } else {
#pragma unroll
        for (int mt = 0; mt < 2; ++mt) {
#pragma unroll
            for (int nt = 0; nt < 4; ++nt) {
                const int nglob = block_n + wn * 64 + nt * 16 + l15;
                const float bv  = bias[nglob];
                const int mbase = block_m + wm * 32 + mt * 16 + lhi * 8;
#pragma unroll
                for (int e = 0; e < 8; ++e) {
                    const int r = mbase + e;
                    const float val = acc[mt][nt][e] + bv;
                    if (MODE == 2) {
                        int b = r >> 12, u = (r >> 6) & 63, t = r & 63;
                        int tok = (b << 12) + (t << 6) + u;
                        ((float*)Out)[(size_t)tok * N + nglob] = val;
                    } else {  // MODE == 3
                        ((float*)Out)[(size_t)r * N + nglob] += val;
                    }
                }
            }
        }
    }
}

// ---------------------------------------------------------------------------
// Axial attention core: one block per (sequence, head). 128 threads = 4 waves.
// S = (Q K^T) * 0.125 + rel_bias ; softmax ; Y = S V     (all 64x64, hd=64)
// ---------------------------------------------------------------------------
__global__ __launch_bounds__(128) void attn_kernel(
        const half_t* __restrict__ Q,
        const half_t* __restrict__ Km,
        const half_t* __restrict__ V,
        const float* __restrict__ relw,
        half_t* __restrict__ Y) {
    const int s = blockIdx.x;   // 0..511
    const int h = blockIdx.y;   // 0..7

    __shared__ half_t qs[64][72];   // Q rows (later reused for softmaxed att)
    __shared__ half_t ks[64][72];   // K rows
    __shared__ half_t vt[64][72];   // V transposed: vt[d][t]
    __shared__ float  sb[64][68];   // raw scores

    const int tid  = threadIdx.x;
    const int lane = tid & 31;
    const int wv   = tid >> 5;
    const int l15  = lane & 15;
    const int lhi  = lane >> 4;
    const size_t base = (size_t)s * TDIM * CDIM + (size_t)h * HD;

    // ---- stage q, k, v(T) in LDS ----
    {
        const int row = tid >> 1;
        const int off = (tid & 1) * 32;
        const half_t* qp = Q  + base + (size_t)row * CDIM + off;
        const half_t* kp = Km + base + (size_t)row * CDIM + off;
        const half_t* vp = V  + base + (size_t)row * CDIM + off;
#pragma unroll
        for (int i = 0; i < 4; ++i) {
            *(float4*)&qs[row][off + i * 8] = *(const float4*)(qp + i * 8);
            *(float4*)&ks[row][off + i * 8] = *(const float4*)(kp + i * 8);
        }
#pragma unroll
        for (int i = 0; i < 32; ++i) vt[off + i][row] = vp[i];
    }
    __syncthreads();

    // ---- S = Q K^T ----
    const int m0 = wv * 16;
    v8f sacc[4];
#pragma unroll
    for (int nt = 0; nt < 4; ++nt) sacc[nt] = vzero8();
#pragma unroll
    for (int kk = 0; kk < 2; ++kk) {
        const half_t* ap = &qs[m0 + l15][kk * 32 + lhi * 8];
        v16h a = ld2(ap, ap + 16);
#pragma unroll
        for (int nt = 0; nt < 4; ++nt) {
            const half_t* bp = &ks[nt * 16 + l15][kk * 32 + lhi * 8];
            v16h b = ld2(bp, bp + 16);
            sacc[nt] = WMMA_F16(a, b, sacc[nt]);
        }
    }
#pragma unroll
    for (int nt = 0; nt < 4; ++nt)
#pragma unroll
        for (int e = 0; e < 8; ++e)
            sb[m0 + lhi * 8 + e][nt * 16 + l15] = sacc[nt][e];
    __syncthreads();

    // ---- softmax row-wise (scale + relative position bias) ----
    if (tid < 64) {
        const int r = tid;
        float mx = -1e30f;
        for (int j = 0; j < 64; ++j) {
            float x = sb[r][j] * 0.125f + relw[(j - r + 63) * NHEAD + h];
            mx = fmaxf(mx, x);
        }
        float sum = 0.f;
        for (int j = 0; j < 64; ++j) {
            float x = sb[r][j] * 0.125f + relw[(j - r + 63) * NHEAD + h];
            sum += __expf(x - mx);
        }
        const float inv = 1.0f / sum;
        for (int j = 0; j < 64; ++j) {
            float x = sb[r][j] * 0.125f + relw[(j - r + 63) * NHEAD + h];
            qs[r][j] = (half_t)(__expf(x - mx) * inv);   // reuse qs as att
        }
    }
    __syncthreads();

    // ---- Y = att @ V ----
    v8f yacc[4];
#pragma unroll
    for (int nt = 0; nt < 4; ++nt) yacc[nt] = vzero8();
#pragma unroll
    for (int kk = 0; kk < 2; ++kk) {
        const half_t* ap = &qs[m0 + l15][kk * 32 + lhi * 8];
        v16h a = ld2(ap, ap + 16);
#pragma unroll
        for (int nt = 0; nt < 4; ++nt) {
            const half_t* bp = &vt[nt * 16 + l15][kk * 32 + lhi * 8];
            v16h b = ld2(bp, bp + 16);
            yacc[nt] = WMMA_F16(a, b, yacc[nt]);
        }
    }
#pragma unroll
    for (int nt = 0; nt < 4; ++nt)
#pragma unroll
        for (int e = 0; e < 8; ++e) {
            const int row = m0 + lhi * 8 + e;
            const int col = nt * 16 + l15;
            Y[base + (size_t)row * CDIM + col] = (half_t)yacc[nt][e];
        }
}

// ---------------------------------------------------------------------------
// out = x + proj   (proj already holds row-proj + col-proj, token layout)
// ---------------------------------------------------------------------------
__global__ void combine_kernel(const float* __restrict__ x,
                               const float* __restrict__ p,
                               float* __restrict__ out, size_t n) {
    size_t i = (size_t)blockIdx.x * blockDim.x + threadIdx.x;
    if (i < n) out[i] = x[i] + p[i];
}

// ---------------------------------------------------------------------------
// Host orchestration
// ---------------------------------------------------------------------------
extern "C" void kernel_launch(void* const* d_in, const int* in_sizes, int n_in,
                              void* d_out, int out_size, void* d_ws, size_t ws_size,
                              hipStream_t stream) {
    (void)in_sizes; (void)n_in; (void)out_size; (void)ws_size;

    const float* x       = (const float*)d_in[0];
    const float* wsrc[8] = { (const float*)d_in[1],  (const float*)d_in[3],
                             (const float*)d_in[5],  (const float*)d_in[7],
                             (const float*)d_in[9],  (const float*)d_in[11],
                             (const float*)d_in[13], (const float*)d_in[15] };
    const float* row_qb  = (const float*)d_in[2];
    const float* row_kb  = (const float*)d_in[4];
    const float* row_vb  = (const float*)d_in[6];
    const float* row_pb  = (const float*)d_in[8];
    const float* col_qb  = (const float*)d_in[10];
    const float* col_kb  = (const float*)d_in[12];
    const float* col_vb  = (const float*)d_in[14];
    const float* col_pb  = (const float*)d_in[16];
    const float* rln1_g  = (const float*)d_in[17];
    const float* rln1_b  = (const float*)d_in[18];
    const float* cln1_g  = (const float*)d_in[19];
    const float* cln1_b  = (const float*)d_in[20];
    const float* ln2_g   = (const float*)d_in[21];
    const float* ln2_b   = (const float*)d_in[22];
    const float* ff_w1   = (const float*)d_in[23];
    const float* ff_b1   = (const float*)d_in[24];
    const float* ff_w2   = (const float*)d_in[25];
    const float* ff_b2   = (const float*)d_in[26];
    const float* row_rel = (const float*)d_in[27];
    const float* col_rel = (const float*)d_in[28];

    // Workspace layout (bytes)
    const size_t EH = (size_t)NTOK * CDIM * sizeof(half_t);  // 33,554,432
    const size_t EF = (size_t)NTOK * CDIM * sizeof(float);   // 67,108,864
    char* ws = (char*)d_ws;
    half_t* buf_ln  = (half_t*)(ws);
    half_t* buf_q   = (half_t*)(ws + EH);
    half_t* buf_k   = (half_t*)(ws + 2 * EH);
    half_t* buf_v   = (half_t*)(ws + 3 * EH);
    half_t* buf_y   = (half_t*)(ws + 4 * EH);
    float*  proj    = (float*)(ws + 5 * EH);
    half_t* buf_hdd = buf_q;                 // reuse q..y region (134 MB)
    float*  out_f   = (float*)d_out;

    // Transposed f16 weights
    char* wbase = ws + 5 * EH + EF;
    const size_t PW = (size_t)CDIM * CDIM * sizeof(half_t);   // 524,288
    half_t* wt[8];
    for (int i = 0; i < 8; ++i) wt[i] = (half_t*)(wbase + i * PW);
    half_t* wt_f1 = (half_t*)(wbase + 8 * PW);                          // [2048][512]
    half_t* wt_f2 = (half_t*)(wbase + 8 * PW + (size_t)CDIM * FFDIM * 2);

    const dim3 bG(256);
    // ---- weight convert/transpose (f32 [K][N] -> f16 [N][K]) ----
    for (int i = 0; i < 8; ++i)
        wcvt_kernel<<<dim3(CDIM / 32, CDIM / 32), bG, 0, stream>>>(wsrc[i], wt[i], CDIM, CDIM);
    wcvt_kernel<<<dim3(CDIM / 32, FFDIM / 32), bG, 0, stream>>>(ff_w1, wt_f1, CDIM, FFDIM);
    wcvt_kernel<<<dim3(FFDIM / 32, CDIM / 32), bG, 0, stream>>>(ff_w2, wt_f2, FFDIM, CDIM);

    const dim3 gLN(NTOK), bLN(128);
    const dim3 gATT(512, NHEAD), bATT(128);
    const dim3 gP(CDIM / 128, NTOK / 128);       // N=512 GEMMs
    const dim3 gF1(FFDIM / 128, NTOK / 128);     // N=2048 GEMM

    // ===================== row attention =====================
    ln_kernel<<<gLN, bLN, 0, stream>>>(x, rln1_g, rln1_b, buf_ln, /*swap=*/1);
    gemm_kernel<0><<<gP, bG, 0, stream>>>(buf_ln, wt[0], row_qb, buf_q, NTOK, CDIM, CDIM);
    gemm_kernel<0><<<gP, bG, 0, stream>>>(buf_ln, wt[1], row_kb, buf_k, NTOK, CDIM, CDIM);
    gemm_kernel<0><<<gP, bG, 0, stream>>>(buf_ln, wt[2], row_vb, buf_v, NTOK, CDIM, CDIM);
    attn_kernel<<<gATT, bATT, 0, stream>>>(buf_q, buf_k, buf_v, row_rel, buf_y);
    gemm_kernel<2><<<gP, bG, 0, stream>>>(buf_y, wt[3], row_pb, proj, NTOK, CDIM, CDIM);

    // ===================== col attention =====================
    ln_kernel<<<gLN, bLN, 0, stream>>>(x, cln1_g, cln1_b, buf_ln, /*swap=*/0);
    gemm_kernel<0><<<gP, bG, 0, stream>>>(buf_ln, wt[4], col_qb, buf_q, NTOK, CDIM, CDIM);
    gemm_kernel<0><<<gP, bG, 0, stream>>>(buf_ln, wt[5], col_kb, buf_k, NTOK, CDIM, CDIM);
    gemm_kernel<0><<<gP, bG, 0, stream>>>(buf_ln, wt[6], col_vb, buf_v, NTOK, CDIM, CDIM);
    attn_kernel<<<gATT, bATT, 0, stream>>>(buf_q, buf_k, buf_v, col_rel, buf_y);
    gemm_kernel<3><<<gP, bG, 0, stream>>>(buf_y, wt[7], col_pb, proj, NTOK, CDIM, CDIM);

    // ===================== residual + FFN =====================
    const size_t nelem = (size_t)NTOK * CDIM;
    combine_kernel<<<dim3((unsigned)((nelem + 255) / 256)), dim3(256), 0, stream>>>(
        x, proj, out_f, nelem);
    ln_kernel<<<gLN, bLN, 0, stream>>>(out_f, ln2_g, ln2_b, buf_ln, /*swap=*/0);
    gemm_kernel<1><<<gF1, bG, 0, stream>>>(buf_ln, wt_f1, ff_b1, buf_hdd, NTOK, FFDIM, CDIM);
    gemm_kernel<3><<<gP, bG, 0, stream>>>(buf_hdd, wt_f2, ff_b2, out_f, NTOK, CDIM, FFDIM);
}
